// Qwen3MoE_1090921693843
// MI455X (gfx1250) — matrix operand order repaired and checked by reading, not compiled, not run on hardware
//
#include <hip/hip_runtime.h>
#include <hip/hip_bf16.h>

// Problem constants (from reference): T tokens, H hidden, I intermediate, E experts, top-K
#define T_TOK 2048
#define HID   1024
#define IDIM  768
#define NEXP  32
#define KTOP  4

typedef __attribute__((ext_vector_type(16))) __bf16 v16bf;
typedef __attribute__((ext_vector_type(8)))  float  v8f;

// ---- fp32 -> bf16 (round-to-nearest-even) and pack helpers ----
__device__ __forceinline__ unsigned short f2bf(float f) {
  unsigned int u = __float_as_uint(f);
  u += 0x7FFFu + ((u >> 16) & 1u);
  return (unsigned short)(u >> 16);
}
__device__ __forceinline__ unsigned int pack_bf2(float lo, float hi) {
  return (unsigned int)f2bf(lo) | ((unsigned int)f2bf(hi) << 16);
}

// ---- gfx1250 async global->LDS copy (ASYNCcnt-tracked DMA path).
// ISA 10: the instruction offset is added to the LDS address and the global address;
//         (offset applies to BOTH sides)
__device__ __forceinline__ void async_copy_b128(const void* gsrc, void* lds) {
  unsigned long long ga = (unsigned long long)(uintptr_t)gsrc;
  unsigned la = (unsigned)(uintptr_t)lds;   // low 32 bits == LDS offset in shared aperture
  asm volatile("global_load_async_to_lds_b128 %0, %1, off"
               :: "v"(la), "v"(ga) : "memory");
}
__device__ __forceinline__ void async_copy_b128_off16(const void* gsrc, void* lds) {
  unsigned long long ga = (unsigned long long)(uintptr_t)gsrc;
  unsigned la = (unsigned)(uintptr_t)lds;
  asm volatile("global_load_async_to_lds_b128 %0, %1, off offset:16"
               :: "v"(la), "v"(ga) : "memory");
}
__device__ __forceinline__ void wait_async0() {
  asm volatile("s_wait_asynccnt 0x0" ::: "memory");
}

// =====================================================================
// Kernel 1: router logits  logits[t][e] = dot(x[t], gate_w[e])
// One lane per (token, expert); gate_w (128 KB) is cache resident.
// =====================================================================
__global__ void router_logits_kernel(const float* __restrict__ x,
                                     const float* __restrict__ gate_w,
                                     float* __restrict__ logits) {
  int gid = blockIdx.x * blockDim.x + threadIdx.x;
  int t = gid >> 5;      // wave-uniform token
  int e = gid & 31;      // lane = expert (E == 32 == wave32)
  const float4* xr = (const float4*)(x + (size_t)t * HID);
  const float4* gr = (const float4*)(gate_w + (size_t)e * HID);
  float acc = 0.f;
#pragma unroll 4
  for (int i = 0; i < HID / 4; ++i) {
    float4 a = xr[i];
    float4 b = gr[i];
    acc += a.x * b.x + a.y * b.y + a.z * b.z + a.w * b.w;
  }
  logits[(size_t)t * NEXP + e] = acc;
}

// =====================================================================
// Kernel 2: softmax + top-4 + renormalize; build per-expert token lists.
// =====================================================================
__global__ void router_topk_kernel(const float* __restrict__ logits,
                                   int* __restrict__ cnt,
                                   int* __restrict__ toks,
                                   float* __restrict__ wts) {
  int t = blockIdx.x * blockDim.x + threadIdx.x;
  if (t >= T_TOK) return;
  float p[NEXP];
  float m = -1e30f;
  for (int e = 0; e < NEXP; ++e) {
    p[e] = logits[(size_t)t * NEXP + e];
    m = fmaxf(m, p[e]);
  }
  float s = 0.f;
  for (int e = 0; e < NEXP; ++e) {
    p[e] = __expf(p[e] - m);
    s += p[e];
  }
  float inv = 1.f / s;
  int   bi[KTOP];
  float bw[KTOP];
  float wsum = 0.f;
  for (int k = 0; k < KTOP; ++k) {
    float best = -1.f;
    int bidx = 0;
    for (int e = 0; e < NEXP; ++e)
      if (p[e] > best) { best = p[e]; bidx = e; }
    bi[k] = bidx;
    bw[k] = best * inv;
    wsum += bw[k];
    p[bidx] = -1.f;
  }
  float rn = 1.f / wsum;
  for (int k = 0; k < KTOP; ++k) {
    int e = bi[k];
    int slot = atomicAdd(&cnt[e], 1);
    toks[(size_t)e * T_TOK + slot] = t;
    wts[(size_t)e * T_TOK + slot] = bw[k] * rn;
  }
}

// =====================================================================
// Kernel 3: one-time fp32 -> packed-bf16 conversion of activations.
// =====================================================================
__global__ void cast_x_kernel(const float* __restrict__ x,
                              unsigned int* __restrict__ xbf) {
  int gid = blockIdx.x * blockDim.x + threadIdx.x;  // 8 floats per thread
  const float4* xp = (const float4*)x;
  float4 a = xp[gid * 2 + 0];
  float4 b = xp[gid * 2 + 1];
  uint4 o = make_uint4(pack_bf2(a.x, a.y), pack_bf2(a.z, a.w),
                       pack_bf2(b.x, b.y), pack_bf2(b.z, b.w));
  ((uint4*)xbf)[gid] = o;
}

// =====================================================================
// Kernel 4: per-(expert, 128-token tile) fused SwiGLU MLP with WMMA bf16.
//   Phase 1: G[128 x 768] = silu(X W1^T) * (X W3^T), kept as bf16 in LDS
//            (192 KB tile -- uses CDNA5's 320 KB WGP LDS).
//   Phase 2: out += diag(w_tok) * (G W2^T) via global_atomic_add_f32.
// 256 threads = 8 wave32; each wave owns a 16-row (token) stripe.
// Activations are staged with GLOBAL_LOAD_ASYNC_TO_LDS_B128 (ASYNCcnt DMA),
// overlapped with the fp32->bf16 VGPR staging of the weight tiles.
// Fragment layouts used here (same contiguous k order for A and B):
//   A 16x32 bf16: lane m=l&15, g=l>>4; VGPR v holds K = 16g+2v,
//                 two packed bf16 per VGPR.
//   B 32x16 bf16: lane n=l&15, g=l>>4; VGPR v holds K = 16g+2v.
//   C/D f32:      VGPR r holds (m = r + 8g, n = l&15).
// =====================================================================
__global__ void __launch_bounds__(256)
expert_mlp_kernel(const unsigned int* __restrict__ xbf,   // [T, H/2] packed bf16
                  const float* __restrict__ w_gate,       // [E, I, H]
                  const float* __restrict__ w_up,         // [E, I, H]
                  const float* __restrict__ w_down,       // [E, H, I]
                  const int*   __restrict__ cnt,
                  const int*   __restrict__ toks,
                  const float* __restrict__ wts,
                  float* __restrict__ out) {              // [T, H]
  const int e    = blockIdx.x / (T_TOK / 128);
  const int tile = blockIdx.x % (T_TOK / 128);
  const int n_e  = cnt[e];
  if (tile * 128 >= n_e) return;   // uniform early exit: empty tile slot

  extern __shared__ char smem_raw[];
  unsigned short* Gs  = (unsigned short*)smem_raw;                 // 128*768 bf16 = 192 KB
  unsigned short* Xs  = Gs  + 128 * IDIM;                          // 128*32 bf16 = 8 KB
  unsigned short* Ws0 = Xs  + 128 * 32;                            // 8 KB (W1 / W2)
  unsigned short* Ws1 = Ws0 + 128 * 32;                            // 8 KB (W3)
  int*   tokid = (int*)(Ws1 + 128 * 32);                           // 512 B
  float* tokw  = (float*)(tokid + 128);                            // 512 B

  const int tid  = threadIdx.x;
  const int wave = tid >> 5;
  const int lane = tid & 31;
  const int lm   = lane & 15;
  const int lg   = lane >> 4;
  const int srow  = tid >> 1;   // staging row 0..127
  const int shalf = tid & 1;    // 16-element half of a 32-wide K chunk

  // Gather this tile's token ids + routing weights (invalid rows -> token 0, weight 0).
  if (tid < 128) {
    int gr = tile * 128 + tid;
    int tkn = 0; float w = 0.f;
    if (gr < n_e) {
      tkn = toks[(size_t)e * T_TOK + gr];
      w   = wts [(size_t)e * T_TOK + gr];
    }
    tokid[tid] = tkn;
    tokw[tid]  = w;
  }
  __syncthreads();

  const size_t wg_base = (size_t)e * IDIM * HID;   // w_gate / w_up expert base
  const v8f vzero = {};

  // ---------------- Phase 1: G = silu(X W1^T) * (X W3^T) ----------------
  for (int ic = 0; ic < IDIM / 128; ++ic) {        // 6 chunks of 128 I-columns
    v8f accg[8], accu[8];
#pragma unroll
    for (int n = 0; n < 8; ++n) { accg[n] = vzero; accu[n] = vzero; }

    for (int kc = 0; kc < HID / 32; ++kc) {        // 32 K-chunks of 32
      // Kick off async DMA of the gathered activation slice (32 B / lane):
      // Xs[srow][shalf*16 .. +16)  <-  xbf[tokid[srow]][kc*32 + shalf*16 ..)
      {
        int tkn = tokid[srow];
        const unsigned int* src = xbf + (size_t)tkn * (HID / 2) + kc * 16 + shalf * 8;
        unsigned short* dst = &Xs[srow * 32 + shalf * 16];
        async_copy_b128(src, dst);          // bytes [0,16)
        async_copy_b128_off16(src, dst);    // bytes [16,32) (offset hits both sides)
      }
      // Meanwhile: stage W1/W3 rows (fp32 -> packed bf16): Ws*[128][32], row = I-col
      {
        const float* b1 = w_gate + wg_base + (size_t)(ic * 128 + srow) * HID + kc * 32 + shalf * 16;
        const float* b3 = w_up   + wg_base + (size_t)(ic * 128 + srow) * HID + kc * 32 + shalf * 16;
        __builtin_prefetch(b1 + 32, 0, 1);   // next K chunk -> global_prefetch_b8
        __builtin_prefetch(b3 + 32, 0, 1);
        float4 a0 = ((const float4*)b1)[0], a1 = ((const float4*)b1)[1];
        float4 a2 = ((const float4*)b1)[2], a3 = ((const float4*)b1)[3];
        *(uint4*)&Ws0[srow * 32 + shalf * 16]     = make_uint4(pack_bf2(a0.x, a0.y), pack_bf2(a0.z, a0.w),
                                                               pack_bf2(a1.x, a1.y), pack_bf2(a1.z, a1.w));
        *(uint4*)&Ws0[srow * 32 + shalf * 16 + 8] = make_uint4(pack_bf2(a2.x, a2.y), pack_bf2(a2.z, a2.w),
                                                               pack_bf2(a3.x, a3.y), pack_bf2(a3.z, a3.w));
        float4 c0 = ((const float4*)b3)[0], c1 = ((const float4*)b3)[1];
        float4 c2 = ((const float4*)b3)[2], c3 = ((const float4*)b3)[3];
        *(uint4*)&Ws1[srow * 32 + shalf * 16]     = make_uint4(pack_bf2(c0.x, c0.y), pack_bf2(c0.z, c0.w),
                                                               pack_bf2(c1.x, c1.y), pack_bf2(c1.z, c1.w));
        *(uint4*)&Ws1[srow * 32 + shalf * 16 + 8] = make_uint4(pack_bf2(c2.x, c2.y), pack_bf2(c2.z, c2.w),
                                                               pack_bf2(c3.x, c3.y), pack_bf2(c3.z, c3.w));
      }
      wait_async0();          // DMA -> LDS complete (ASYNCcnt == 0)
      __syncthreads();

      // A fragment: this wave's 16 token rows.
      union { v16bf v; unsigned int u[8]; } af;
      const unsigned int* Xu = (const unsigned int*)Xs;
      const int arow = wave * 16 + lm;
#pragma unroll
      for (int v = 0; v < 8; ++v) {
        int kb = 16 * lg + 2 * v;
        af.u[v] = Xu[arow * 16 + (kb >> 1)];
      }
      const unsigned int* W0u = (const unsigned int*)Ws0;
      const unsigned int* W1u = (const unsigned int*)Ws1;
#pragma unroll
      for (int ns = 0; ns < 8; ++ns) {
        const int brow = ns * 16 + lm;
        union { v16bf v; unsigned int u[8]; } bg, bu;
#pragma unroll
        for (int v = 0; v < 8; ++v) {
          int idx = brow * 16 + 8 * lg + v;   // B layout: K = 16g + 2v
          bg.u[v] = W0u[idx];
          bu.u[v] = W1u[idx];
        }
        accg[ns] = __builtin_amdgcn_wmma_f32_16x16x32_bf16(false, af.v, false, bg.v,
                                                           (short)0, accg[ns], false, false);
        accu[ns] = __builtin_amdgcn_wmma_f32_16x16x32_bf16(false, af.v, false, bu.v,
                                                           (short)0, accu[ns], false, false);
      }
      __syncthreads();
    }

    // SwiGLU epilogue -> bf16 into the big LDS G tile.
#pragma unroll
    for (int ns = 0; ns < 8; ++ns) {
#pragma unroll
      for (int r = 0; r < 8; ++r) {
        float g = accg[ns][r];
        float u = accu[ns][r];
        float h = (g / (1.f + __expf(-g))) * u;
        int m   = wave * 16 + r + 8 * lg;          // token row (C/D layout)
        int col = ic * 128 + ns * 16 + lm;         // I column
        Gs[m * IDIM + col] = f2bf(h);
      }
    }
  }
  __syncthreads();

  // ---------------- Phase 2: out += diag(w) * (G W2^T) ----------------
  const size_t wd_base = (size_t)e * HID * IDIM;
  for (int hc = 0; hc < HID / 128; ++hc) {         // 8 chunks of 128 H-columns
    v8f acco[8];
#pragma unroll
    for (int n = 0; n < 8; ++n) acco[n] = vzero;

    for (int kc = 0; kc < IDIM / 32; ++kc) {       // 24 K-chunks over I
      // Stage W2 rows (fp32 -> bf16): Ws0[128][32], row = H-column
      {
        const float* b2 = w_down + wd_base + (size_t)(hc * 128 + srow) * IDIM + kc * 32 + shalf * 16;
        __builtin_prefetch(b2 + 32, 0, 1);
        float4 a0 = ((const float4*)b2)[0], a1 = ((const float4*)b2)[1];
        float4 a2 = ((const float4*)b2)[2], a3 = ((const float4*)b2)[3];
        *(uint4*)&Ws0[srow * 32 + shalf * 16]     = make_uint4(pack_bf2(a0.x, a0.y), pack_bf2(a0.z, a0.w),
                                                               pack_bf2(a1.x, a1.y), pack_bf2(a1.z, a1.w));
        *(uint4*)&Ws0[srow * 32 + shalf * 16 + 8] = make_uint4(pack_bf2(a2.x, a2.y), pack_bf2(a2.z, a2.w),
                                                               pack_bf2(a3.x, a3.y), pack_bf2(a3.z, a3.w));
      }
      __syncthreads();

      union { v16bf v; unsigned int u[8]; } af;
      const unsigned int* Gu = (const unsigned int*)Gs;
      const int arow = wave * 16 + lm;
#pragma unroll
      for (int v = 0; v < 8; ++v) {
        int kb = 16 * lg + 2 * v;
        af.u[v] = Gu[arow * (IDIM / 2) + ((kc * 32 + kb) >> 1)];
      }
      const unsigned int* W0u = (const unsigned int*)Ws0;
#pragma unroll
      for (int ns = 0; ns < 8; ++ns) {
        const int brow = ns * 16 + lm;
        union { v16bf v; unsigned int u[8]; } b2;
#pragma unroll
        for (int v = 0; v < 8; ++v) b2.u[v] = W0u[brow * 16 + 8 * lg + v];
        acco[ns] = __builtin_amdgcn_wmma_f32_16x16x32_bf16(false, af.v, false, b2.v,
                                                           (short)0, acco[ns], false, false);
      }
      __syncthreads();
    }

    // Weighted scatter: multiple experts hit the same token -> f32 atomics.
#pragma unroll
    for (int ns = 0; ns < 8; ++ns) {
#pragma unroll
      for (int r = 0; r < 8; ++r) {
        int m = wave * 16 + r + 8 * lg;
        int tkn = tokid[m];
        float w = tokw[m];                          // 0 for padded rows
        int hcol = hc * 128 + ns * 16 + lm;
        atomicAdd(&out[(size_t)tkn * HID + hcol], w * acco[ns][r]);
      }
    }
  }
}

// =====================================================================
// Host-side launch (graph-capture safe: only async stream ops).
// =====================================================================
extern "C" void kernel_launch(void* const* d_in, const int* in_sizes, int n_in,
                              void* d_out, int out_size, void* d_ws, size_t ws_size,
                              hipStream_t stream) {
  const float* x      = (const float*)d_in[0];
  const float* gate_w = (const float*)d_in[1];
  const float* w_gate = (const float*)d_in[2];
  const float* w_up   = (const float*)d_in[3];
  const float* w_down = (const float*)d_in[4];
  float* out = (float*)d_out;

  // Workspace layout (bytes):
  //   [0)          logits  T*E*4   = 262144
  //   [262144)     cnt     E*4     = 128
  //   [262272)     toks    E*T*4   = 262144
  //   [524416)     wts     E*T*4   = 262144
  //   [786560)     xbf     T*H*2   = 4194304      (packed bf16)
  char* ws = (char*)d_ws;
  float*        logits = (float*)(ws + 0);
  int*          cnt    = (int*)  (ws + 262144);
  int*          toks   = (int*)  (ws + 262272);
  float*        wts    = (float*)(ws + 524416);
  unsigned int* xbf    = (unsigned int*)(ws + 786560);

  hipMemsetAsync(out, 0, (size_t)T_TOK * HID * sizeof(float), stream);
  hipMemsetAsync(cnt, 0, NEXP * sizeof(int), stream);

  router_logits_kernel<<<(T_TOK * NEXP) / 256, 256, 0, stream>>>(x, gate_w, logits);
  router_topk_kernel<<<T_TOK / 256, 256, 0, stream>>>(logits, cnt, toks, wts);
  cast_x_kernel<<<(T_TOK * HID) / 8 / 256, 256, 0, stream>>>(x, xbf);

  // LDS: G tile 192 KB + 3x 8 KB staging + token id/weight arrays = 217 KB
  size_t smem = (size_t)128 * IDIM * 2 + 3 * 128 * 32 * 2 + 128 * 4 + 128 * 4;
  expert_mlp_kernel<<<NEXP * (T_TOK / 128), 256, smem, stream>>>(
      xbf, w_gate, w_up, w_down, cnt, toks, wts, out);
}
